// Retention_12601434046818
// MI455X (gfx1250) — matrix operand order repaired and checked by reading, not compiled, not run on hardware
//
#include <hip/hip_runtime.h>
#include <hip/hip_bf16.h>

#define HEAD  128
#define SEQ   4096
#define BATCH 4
#define EMB   1024
// log2(0.96875)
#define L2GAMMA (-0.045803689611586f)
// log2(10000)
#define L2_10K  (13.287712379549449f)

typedef __bf16 bf16_t;
typedef __attribute__((ext_vector_type(8)))  __bf16 bf16x8;   // 128-bit
typedef __attribute__((ext_vector_type(16))) __bf16 v16bf;
typedef __attribute__((ext_vector_type(8)))  float  v8f;
typedef __attribute__((ext_vector_type(4)))  float  f32x4;

static __device__ __forceinline__ v16bf pack16(bf16x8 lo, bf16x8 hi) {
  v16bf r;
#pragma unroll
  for (int i = 0; i < 8; ++i) { r[i] = lo[i]; r[i + 8] = hi[i]; }
  return r;
}

// load 8 consecutive f32 (two b128 loads), convert to bf16x8
static __device__ __forceinline__ bf16x8 cvt8(const float* p) {
  f32x4 a = *(const f32x4*)p;
  f32x4 b = *(const f32x4*)(p + 4);
  bf16x8 r;
#pragma unroll
  for (int i = 0; i < 4; ++i) { r[i] = (__bf16)a[i]; r[i + 4] = (__bf16)b[i]; }
  return r;
}

// ---------------------------------------------------------------------------
// Kernel 1: convert the three weight matrices f32 -> bf16 (one-time, 1.5 MB).
// ---------------------------------------------------------------------------
__global__ void cvt_w_kernel(const float* __restrict__ Wq,
                             const float* __restrict__ Wk,
                             const float* __restrict__ Wv,
                             bf16_t* __restrict__ Wb) {
  const int N = HEAD * EMB;
  int idx = blockIdx.x * blockDim.x + threadIdx.x;
  if (idx < N) {
    Wb[idx]         = (__bf16)Wq[idx];
    Wb[N + idx]     = (__bf16)Wk[idx];
    Wb[2 * N + idx] = (__bf16)Wv[idx];
  }
}

// ---------------------------------------------------------------------------
// Kernel 2: projection GEMM (M=B*S, N=H, K=E) + bias + xPos rotary, fused.
// mode: 0 = Q (xpos upscale), 1 = K (xpos downscale), 2 = V (transposed store)
// One wave per 16-row tile; 8 WMMA accumulators cover all H=128 columns.
// ---------------------------------------------------------------------------
__global__ __launch_bounds__(128)
void proj_xpos_kernel(const float* __restrict__ x,
                      const bf16_t* __restrict__ Wb,
                      const float* __restrict__ bias,
                      bf16_t* __restrict__ dst,
                      int mode) {
  const int lane = threadIdx.x & 31;
  const int wave = threadIdx.x >> 5;
  const int l15  = lane & 15;
  const int hi   = lane >> 4;       // half-lane select
  const int koff = hi * 8;          // A-matrix interleaved K offset
  const int boff = hi * 8;          // B-matrix interleaved K offset

  const int R0   = (blockIdx.x * 4 + wave) * 16;   // global row base (< B*S)
  const int arow = R0 + l15;

  v8f acc[8];
#pragma unroll
  for (int t = 0; t < 8; ++t) acc[t] = (v8f)(0.0f);

  const float* xrow = x + (size_t)arow * EMB;
  for (int kk = 0; kk < EMB; kk += 32) {
    v16bf A = pack16(cvt8(xrow + kk + koff), cvt8(xrow + kk + 16 + koff));
#pragma unroll
    for (int t = 0; t < 8; ++t) {
      const bf16_t* wrow = Wb + (size_t)(t * 16 + l15) * EMB + kk;
      v16bf Bm = pack16(*(const bf16x8*)(wrow + boff),
                        *(const bf16x8*)(wrow + boff + 16));
      acc[t] = __builtin_amdgcn_wmma_f32_16x16x32_bf16(
          false, A, false, Bm, (short)0, acc[t], false, false);
    }
  }

  // C/D layout: lane holds column h = t*16 + l15; VGPR r holds row r + 8*hi.
  const int sBase = (R0 % SEQ) + 8 * hi;
#pragma unroll
  for (int t = 0; t < 8; ++t) {
    const int h  = t * 16 + l15;
    const float bv = bias[h];
    v8f a = acc[t];
#pragma unroll
    for (int r = 0; r < 8; ++r) a[r] += bv;

    if (mode < 2) {
      // xPos: per lane h is fixed -> freq/scale-base are loop invariant
      const int   i    = h >> 1;
      const float invf = exp2f(-(float)i * (L2_10K / 64.0f));
      float       lsb  = log2f(((float)(2 * i) + 51.2f) / 179.2f);
      if (mode == 1) lsb = -lsb;                 // downscale for K
      const float sgn  = (h & 1) ? 1.0f : -1.0f; // rotate-every-two sign
#pragma unroll
      for (int r = 0; r < 8; ++r) {
        float xv  = a[r];
        float pr  = __shfl_xor(xv, 1, 32);       // partner of the (even,odd) pair
        float rot = sgn * pr;
        float s   = (float)(sBase + r);
        float sc  = exp2f(lsb * (s * (1.0f / 512.0f)));
        float ang = s * invf;
        a[r] = (xv * __cosf(ang) + rot * __sinf(ang)) * sc;
      }
    }

    if (mode == 2) {
      // V stored transposed: Vt[h][b*S+m] so attention GEMM2 B-loads are contiguous
#pragma unroll
      for (int r = 0; r < 8; ++r)
        dst[(size_t)h * (BATCH * SEQ) + (R0 + r + 8 * hi)] = (__bf16)a[r];
    } else {
#pragma unroll
      for (int r = 0; r < 8; ++r)
        dst[(size_t)(R0 + r + 8 * hi) * HEAD + h] = (__bf16)a[r];
    }
  }
}

// ---------------------------------------------------------------------------
// Kernel 3: retention attention. One wave = 16 queries x full H=128.
// Loops 32-key chunks up to the diagonal: GEMM1 (Q.K^T, 8 wmma), decay in f32,
// LDS transpose of the score tile into A-layout, GEMM2 (scores.V, 8 wmma).
// ---------------------------------------------------------------------------
__global__ __launch_bounds__(128)
void retention_attn_kernel(const bf16_t* __restrict__ Qb,
                           const bf16_t* __restrict__ Kb,
                           const bf16_t* __restrict__ Vt,
                           float* __restrict__ out) {
  __shared__ bf16_t smem[4][16][32];   // per-wave 16(n) x 32(m) score staging

  const int lane = threadIdx.x & 31;
  const int wave = threadIdx.x >> 5;
  const int l15  = lane & 15;
  const int hi   = lane >> 4;
  const int koff = hi * 8;            // A-layout interleaved K offset
  const int boff = hi * 8;            // B-layout interleaved K offset

  const int    tilesPerSeq = SEQ / 64;
  const int    b    = blockIdx.x / tilesPerSeq;
  const int    N0   = (blockIdx.x % tilesPerSeq) * 64 + wave * 16;
  const size_t base = (size_t)b * SEQ;

  // Preload the wave's Q A-tiles (16 rows x H=128 -> 4 tiles of 16x32)
  v16bf Qa[4];
  {
    const bf16_t* qrow = Qb + (base + N0 + l15) * HEAD;
#pragma unroll
    for (int t = 0; t < 4; ++t)
      Qa[t] = pack16(*(const bf16x8*)(qrow + t * 32 + koff),
                     *(const bf16x8*)(qrow + t * 32 + 16 + koff));
  }

  v8f o[8];
#pragma unroll
  for (int t = 0; t < 8; ++t) o[t] = (v8f)(0.0f);

  const int Nend = N0 + 15;
  for (int M0 = 0; M0 <= Nend; M0 += 32) {
    if (M0 + 32 <= Nend)  // prefetch next K chunk (global_prefetch_b8)
      __builtin_prefetch(Kb + (base + M0 + 32 + l15) * HEAD, 0, 1);

    // ---- GEMM1: scores for two 16-key subtiles
#pragma unroll
    for (int t2 = 0; t2 < 2; ++t2) {
      const int m = M0 + t2 * 16 + l15;       // this lane's key (B-layout col)
      const bf16_t* krow = Kb + (base + m) * HEAD;
      v8f sc = (v8f)(0.0f);
#pragma unroll
      for (int t = 0; t < 4; ++t) {
        v16bf Bm = pack16(*(const bf16x8*)(krow + t * 32 + boff),
                          *(const bf16x8*)(krow + t * 32 + boff + 16));
        sc = __builtin_amdgcn_wmma_f32_16x16x32_bf16(
            false, Qa[t], false, Bm, (short)0, sc, false, false);
      }
      // decay weight gamma^(n-m), causal mask, stage to LDS as [n][m] bf16
#pragma unroll
      for (int r = 0; r < 8; ++r) {
        const int n    = N0 + r + 8 * hi;
        const int diff = n - m;
        const float w  = (diff >= 0) ? exp2f((float)diff * L2GAMMA) : 0.0f;
        smem[wave][r + 8 * hi][t2 * 16 + l15] = (__bf16)(sc[r] * w);
      }
    }
    // cross-lane LDS dependency within the wave: wait for all DS stores
    asm volatile("s_wait_dscnt 0x0" ::: "memory");

    // ---- reload decayed scores in A-matrix layout (16 n-rows x 32 m-K)
    v16bf sA;
    {
      const bf16_t* srow = &smem[wave][l15][0];
      sA = pack16(*(const bf16x8*)(srow + koff),
                  *(const bf16x8*)(srow + 16 + koff));
    }

    // ---- GEMM2: out += scores x V (V transposed: per-lane contiguous m)
#pragma unroll
    for (int t = 0; t < 8; ++t) {
      const bf16_t* vcol = Vt + (size_t)(t * 16 + l15) * (BATCH * SEQ) + base + M0;
      v16bf Bm = pack16(*(const bf16x8*)(vcol + boff),
                        *(const bf16x8*)(vcol + boff + 16));
      o[t] = __builtin_amdgcn_wmma_f32_16x16x32_bf16(
          false, sA, false, Bm, (short)0, o[t], false, false);
    }
  }

  // ---- store f32 output tile: lane = column h, VGPR r = row n
#pragma unroll
  for (int t = 0; t < 8; ++t) {
    const int h = t * 16 + l15;
#pragma unroll
    for (int r = 0; r < 8; ++r)
      out[(base + N0 + r + 8 * hi) * HEAD + h] = o[t][r];
  }
}

// ---------------------------------------------------------------------------
extern "C" void kernel_launch(void* const* d_in, const int* in_sizes, int n_in,
                              void* d_out, int out_size, void* d_ws, size_t ws_size,
                              hipStream_t stream) {
  const float* q  = (const float*)d_in[0];
  const float* k  = (const float*)d_in[1];
  const float* v  = (const float*)d_in[2];
  const float* Wq = (const float*)d_in[3];
  const float* bq = (const float*)d_in[4];
  const float* Wk = (const float*)d_in[5];
  const float* bk = (const float*)d_in[6];
  const float* Wv = (const float*)d_in[7];
  const float* bv = (const float*)d_in[8];
  float* out = (float*)d_out;

  const size_t NW = (size_t)HEAD * EMB;          // 131072 elems per W
  const size_t NT = (size_t)BATCH * SEQ * HEAD;  // 2M elems per tensor
  bf16_t* Wqb = (bf16_t*)d_ws;
  bf16_t* Wkb = Wqb + NW;
  bf16_t* Wvb = Wkb + NW;
  bf16_t* Qbf = Wvb + NW;
  bf16_t* Kbf = Qbf + NT;
  bf16_t* Vtb = Kbf + NT;   // transposed layout H x (B*S)

  // 1) W -> bf16
  cvt_w_kernel<<<(HEAD * EMB + 255) / 256, 256, 0, stream>>>(Wq, Wk, Wv, Wqb);

  // 2) projections + xPos (Q up, K down, V transposed)
  const int projBlocks = (BATCH * SEQ) / 64;     // 256 blocks of 4 waves
  proj_xpos_kernel<<<projBlocks, 128, 0, stream>>>(q, Wqb, bq, Qbf, 0);
  proj_xpos_kernel<<<projBlocks, 128, 0, stream>>>(k, Wkb, bk, Kbf, 1);
  proj_xpos_kernel<<<projBlocks, 128, 0, stream>>>(v, Wvb, bv, Vtb, 2);

  // 3) retention attention
  retention_attn_kernel<<<(BATCH * SEQ) / 64, 128, 0, stream>>>(Qbf, Kbf, Vtb, out);
}
